// BiLSTM_CRF_11957188952419
// MI455X (gfx1250) — compile-verified, matrix operand order checked by reading, not run on hardware
//
#include <hip/hip_runtime.h>
#include <hip/hip_bf16.h>

// ---------------------------------------------------------------------------
// BiLSTM-CRF for MI455X (gfx1250, wave32).
//   L=512, EMB=512, HID2=512, 4*HID2=2048, T=50 (START=48, STOP=49)
// Pipeline:
//   1) gather_embed_f16 : x = embed[sentence] -> f16              (512x512)
//   2) prep_weights     : Wih->f16, W_out->f16 (padded 64), bias  folds
//   3) gemm_xg (WMMA)   : xg = x @ Wih^T + bias; async-LDS A tile (2x512x2048)
//   4) lstm_kernel      : sequential recurrence, h/c in LDS, f16 h out
//   5) feats_wmma       : feats = hcat @ W_out^T + b_out          (512x64)
//   6) viterbi_kernel   : max-scan + backtrack                    (1 block)
// ---------------------------------------------------------------------------

typedef _Float16 v16h __attribute__((ext_vector_type(16)));
typedef float    v8f  __attribute__((ext_vector_type(8)));

union FragH { v16h v; uint4 q[2]; };

#define L_SENT 512
#define EMB    512
#define H2     512
#define G4     2048   // 4*H2
#define NTAG   50
#define NTPAD  64     // padded tag dim for WMMA
#define TAG_START 48
#define TAG_STOP  49

// ---- 1) embedding gather + fp32->f16 convert ------------------------------
__global__ __launch_bounds__(256) void gather_embed_f16(
    const int* __restrict__ sentence, const float* __restrict__ embed,
    _Float16* __restrict__ xh)
{
  int i = blockIdx.x * 256 + threadIdx.x;
  if (i < L_SENT * EMB) {
    int t = i >> 9;           // / EMB
    int e = i & (EMB - 1);
    xh[i] = (_Float16)embed[(size_t)sentence[t] * EMB + e];
  }
}

// ---- 2) weight converts + bias fold ---------------------------------------
__global__ __launch_bounds__(256) void prep_weights(
    const float* __restrict__ Wih_f, const float* __restrict__ Wih_b,
    const float* __restrict__ bih_f, const float* __restrict__ bhh_f,
    const float* __restrict__ bih_b, const float* __restrict__ bhh_b,
    const float* __restrict__ W_out,
    _Float16* __restrict__ wh, float* __restrict__ bsum,
    _Float16* __restrict__ woh)        // (64, 1024) zero-padded
{
  int i = blockIdx.x * 256 + threadIdx.x;
  const int NW = G4 * EMB;
  if (i < NW) {
    wh[i]      = (_Float16)Wih_f[i];
    wh[NW + i] = (_Float16)Wih_b[i];
  }
  if (i < G4) {
    bsum[i]      = bih_f[i] + bhh_f[i];
    bsum[G4 + i] = bih_b[i] + bhh_b[i];
  }
  if (i < NTPAD * 1024) {
    int row = i >> 10, col = i & 1023;
    woh[i] = (row < NTAG) ? (_Float16)W_out[row * 1024 + col] : (_Float16)0.0f;
  }
}

// ---- 3) xg GEMM via v_wmma_f32_16x16x32_f16, async-LDS staged A -----------
// 256-thread block = 8 waves. Block tile: 16 (M=time) x 512 (N=gate).
// The shared 16x512 f16 A tile (16 KB) is staged into LDS once with
// global_load_async_to_lds_b128 (ASYNCcnt), then each wave does N=4 register
// blocking. All 4 B fragments of a K-step are loaded before any WMMA so the
// 8 global_load_b128 stay in flight together (min-1-block launch bounds give
// the allocator room for the extra live fragments).
__global__ __launch_bounds__(256, 1) void gemm_xg_kernel(
    const _Float16* __restrict__ xh,   // (512, 512)
    const _Float16* __restrict__ wh,   // (2, 2048, 512)
    const float* __restrict__ bsum,    // (2, 2048)
    float* __restrict__ xg)            // (2, 512, 2048)
{
  __shared__ __align__(16) _Float16 atile[16 * EMB];   // 16 KB

  const int dir  = blockIdx.z;
  const int m0   = blockIdx.y * 16;                    // timestep tile
  const int tid  = threadIdx.x;
  const int lane = tid & 31;
  const int wave = tid >> 5;
  const int r    = lane & 15;
  const int hi   = lane >> 4;

  // --- stage A tile: 16 rows x 1024 B; 64 x b128 segments per row ---------
  {
    const unsigned lds_base = (unsigned)(uintptr_t)(&atile[0]);
#pragma unroll
    for (int it = 0; it < 4; ++it) {
      const int idx = it * 256 + tid;                  // 0..1023
      const int row = idx >> 6;
      const int seg = idx & 63;
      const unsigned long long g =
          (unsigned long long)(uintptr_t)(xh + (size_t)(m0 + row) * EMB + seg * 8);
      const unsigned l = lds_base + (unsigned)(row * (EMB * 2) + seg * 16);
      asm volatile("global_load_async_to_lds_b128 %0, %1, off"
                   :: "v"(l), "v"(g) : "memory");
    }
    asm volatile("s_wait_asynccnt 0x0" ::: "memory");
  }
  __syncthreads();

  const int n_base = blockIdx.x * 512 + wave * 64;     // this wave's 4 tiles
  const _Float16* wd = wh + (size_t)dir * G4 * EMB;
  const _Float16* arow_l = atile + (size_t)r * EMB;

  // hoisted B row pointers: K-loop addresses via instruction offsets only
  const _Float16* brow0 = wd + (size_t)(n_base +  0 + r) * EMB;
  const _Float16* brow1 = wd + (size_t)(n_base + 16 + r) * EMB;
  const _Float16* brow2 = wd + (size_t)(n_base + 32 + r) * EMB;
  const _Float16* brow3 = wd + (size_t)(n_base + 48 + r) * EMB;
  const _Float16* brow[4] = { brow0, brow1, brow2, brow3 };

  v8f acc[4] = {};
#pragma unroll
  for (int k0 = 0; k0 < EMB; k0 += 32) {
    const int base = k0 + hi * 8;
    FragH a;
    a.q[0] = *(const uint4*)(arow_l + base);           // ds_load_b128
    a.q[1] = *(const uint4*)(arow_l + base + 16);
    FragH b[4];
#pragma unroll
    for (int tt = 0; tt < 4; ++tt) {                   // issue all 8 b128 loads
      b[tt].q[0] = *(const uint4*)(brow[tt] + base);
      b[tt].q[1] = *(const uint4*)(brow[tt] + base + 16);
    }
#pragma unroll
    for (int tt = 0; tt < 4; ++tt) {                   // then 4 WMMAs
      acc[tt] = __builtin_amdgcn_wmma_f32_16x16x32_f16(
          false, a.v, false, b[tt].v, (short)0, acc[tt], false, false);
    }
  }

  float* outd = xg + (size_t)dir * L_SENT * G4;
#pragma unroll
  for (int tt = 0; tt < 4; ++tt) {
    const int n = n_base + tt * 16 + r;
    const float bias = bsum[dir * G4 + n];
#pragma unroll
    for (int j = 0; j < 8; ++j) {
      const int m = m0 + j + 8 * hi;                   // D layout: M=j+8*hi, N=r
      outd[(size_t)m * G4 + n] = acc[tt][j] + bias;
    }
  }
}

// ---- 4) sequential LSTM recurrence (one block per direction) --------------
// 512 threads: thread j owns hidden unit j; 4 gate dots against h (LDS),
// vectorized float4 (b128 LDS + global loads). Whh L2-resident.
__global__ __launch_bounds__(512) void lstm_kernel(
    const float* __restrict__ xg,     // (2, 512, 2048)
    const float* __restrict__ Whh_f,  // (2048, 512)
    const float* __restrict__ Whh_b,  // (2048, 512)
    _Float16* __restrict__ hcat_h)    // (512, 1024) f16
{
  const int dir = blockIdx.x;
  const int j   = threadIdx.x;        // 0..511
  const float* Whh = dir ? Whh_b : Whh_f;
  const float* xgd = xg + (size_t)dir * L_SENT * G4;

  __shared__ __align__(16) float h[H2];
  __shared__ float c[H2];
  h[j] = 0.0f; c[j] = 0.0f;
  __syncthreads();

  const float4* Wi4 = (const float4*)(Whh + (size_t)(0 * H2 + j) * H2);
  const float4* Wf4 = (const float4*)(Whh + (size_t)(1 * H2 + j) * H2);
  const float4* Wg4 = (const float4*)(Whh + (size_t)(2 * H2 + j) * H2);
  const float4* Wo4 = (const float4*)(Whh + (size_t)(3 * H2 + j) * H2);
  const float4* h4  = (const float4*)h;

  for (int s = 0; s < L_SENT; ++s) {
    const int t = dir ? (L_SENT - 1 - s) : s;
    const float* xr = xgd + (size_t)t * G4;
    float gi = xr[j], gf = xr[H2 + j], gg = xr[2 * H2 + j], go = xr[3 * H2 + j];

    // prefetch next timestep's xg row (global_prefetch_b8)
    const int tn = dir ? (t - 1) : (t + 1);
    if (tn >= 0 && tn < L_SENT)
      __builtin_prefetch(xgd + (size_t)tn * G4 + j, 0, 1);

#pragma unroll 2
    for (int k = 0; k < H2 / 4; ++k) {
      const float4 hv = h4[k];
      const float4 wi = Wi4[k], wf = Wf4[k], wg = Wg4[k], wo = Wo4[k];
      gi = fmaf(wi.x, hv.x, gi); gi = fmaf(wi.y, hv.y, gi);
      gi = fmaf(wi.z, hv.z, gi); gi = fmaf(wi.w, hv.w, gi);
      gf = fmaf(wf.x, hv.x, gf); gf = fmaf(wf.y, hv.y, gf);
      gf = fmaf(wf.z, hv.z, gf); gf = fmaf(wf.w, hv.w, gf);
      gg = fmaf(wg.x, hv.x, gg); gg = fmaf(wg.y, hv.y, gg);
      gg = fmaf(wg.z, hv.z, gg); gg = fmaf(wg.w, hv.w, gg);
      go = fmaf(wo.x, hv.x, go); go = fmaf(wo.y, hv.y, go);
      go = fmaf(wo.z, hv.z, go); go = fmaf(wo.w, hv.w, go);
    }

    const float si = 1.0f / (1.0f + __expf(-gi));
    const float sf = 1.0f / (1.0f + __expf(-gf));
    const float so = 1.0f / (1.0f + __expf(-go));
    const float cn = sf * c[j] + si * tanhf(gg);
    const float hn = so * tanhf(cn);

    __syncthreads();                 // all gate reads of h done
    h[j] = hn; c[j] = cn;
    hcat_h[(size_t)t * 1024 + dir * H2 + j] = (_Float16)hn;
    __syncthreads();                 // h updated before next step's reads
  }
}

// ---- 5) feats = hcat @ W_out^T + b_out via WMMA ---------------------------
// One wave per 16x16 tile of the (512 x 64-padded) output; K = 1024.
__global__ __launch_bounds__(32) void feats_wmma_kernel(
    const _Float16* __restrict__ hcat_h,  // (512, 1024)
    const _Float16* __restrict__ woh,     // (64, 1024) zero-padded
    const float* __restrict__ b_out,      // (50)
    float* __restrict__ feats)            // (512, 64)
{
  const int n0   = blockIdx.x * 16;
  const int m0   = blockIdx.y * 16;
  const int lane = threadIdx.x;
  const int r    = lane & 15;
  const int hi   = lane >> 4;

  const _Float16* arow = hcat_h + (size_t)(m0 + r) * 1024;
  const _Float16* brow = woh    + (size_t)(n0 + r) * 1024;

  v8f acc = {};
#pragma unroll
  for (int k0 = 0; k0 < 1024; k0 += 32) {
    const int base = k0 + hi * 8;
    FragH a, b;
    a.q[0] = *(const uint4*)(arow + base);
    a.q[1] = *(const uint4*)(arow + base + 16);
    b.q[0] = *(const uint4*)(brow + base);
    b.q[1] = *(const uint4*)(brow + base + 16);
    acc = __builtin_amdgcn_wmma_f32_16x16x32_f16(
        false, a.v, false, b.v, (short)0, acc, false, false);
  }

  const int n = n0 + r;
  const float bias = (n < NTAG) ? b_out[n] : 0.0f;
#pragma unroll
  for (int j = 0; j < 8; ++j) {
    const int m = m0 + j + 8 * hi;
    feats[(size_t)m * NTPAD + n] = acc[j] + bias;
  }
}

// ---- 6) Viterbi decode + backtrack ----------------------------------------
__global__ __launch_bounds__(64) void viterbi_kernel(
    const float* __restrict__ feats,      // (512, 64) padded
    const float* __restrict__ trans,
    int* __restrict__ bp, float* __restrict__ out)
{
  __shared__ float fv[NTAG];
  __shared__ float fvn[NTAG];
  __shared__ float tr[NTAG * NTAG];
  const int tid = threadIdx.x;
  for (int i = tid; i < NTAG * NTAG; i += 64) tr[i] = trans[i];
  if (tid < NTAG) fv[tid] = (tid == TAG_START) ? 0.0f : -10000.0f;
  __syncthreads();

  for (int t = 0; t < L_SENT; ++t) {
    if (tid < NTAG) {
      const float* trow = tr + tid * NTAG;   // transitions[next][prev]
      float best = -3.4e38f; int bi = 0;
      for (int p = 0; p < NTAG; ++p) {
        const float s = fv[p] + trow[p];
        if (s > best) { best = s; bi = p; }
      }
      fvn[tid] = best + feats[t * NTPAD + tid];
      bp[t * NTAG + tid] = bi;
    }
    __syncthreads();
    if (tid < NTAG) fv[tid] = fvn[tid];
    __syncthreads();
  }

  if (tid == 0) {
    float best = -3.4e38f; int bi = 0;
    for (int p = 0; p < NTAG; ++p) {
      const float s = fv[p] + tr[TAG_STOP * NTAG + p];
      if (s > best) { best = s; bi = p; }
    }
    out[0] = best;                    // path_score
    int tag = bi;
    for (int t = L_SENT - 1; t >= 0; --t) {
      out[1 + t] = (float)tag;        // path[t]
      tag = bp[t * NTAG + tag];
    }
  }
}

// ---------------------------------------------------------------------------
extern "C" void kernel_launch(void* const* d_in, const int* in_sizes, int n_in,
                              void* d_out, int out_size, void* d_ws, size_t ws_size,
                              hipStream_t stream) {
  const int*   sentence = (const int*)  d_in[0];
  const float* embed    = (const float*)d_in[1];
  const float* Wih_f    = (const float*)d_in[2];
  const float* Whh_f    = (const float*)d_in[3];
  const float* bih_f    = (const float*)d_in[4];
  const float* bhh_f    = (const float*)d_in[5];
  const float* Wih_b    = (const float*)d_in[6];
  const float* Whh_b    = (const float*)d_in[7];
  const float* bih_b    = (const float*)d_in[8];
  const float* bhh_b    = (const float*)d_in[9];
  const float* W_out    = (const float*)d_in[10];
  const float* b_out    = (const float*)d_in[11];
  const float* trans    = (const float*)d_in[12];

  char* ws = (char*)d_ws;
  size_t off = 0;
  auto alloc = [&](size_t bytes) -> void* {
    void* p = ws + off;
    off = (off + bytes + 255) & ~(size_t)255;
    return p;
  };
  _Float16* xh     = (_Float16*)alloc((size_t)L_SENT * EMB * 2);        // 512 KB
  _Float16* wh     = (_Float16*)alloc((size_t)2 * G4 * EMB * 2);        // 4 MB
  float*    bsum   = (float*)   alloc((size_t)2 * G4 * 4);
  _Float16* woh    = (_Float16*)alloc((size_t)NTPAD * 1024 * 2);        // 128 KB
  float*    xg     = (float*)   alloc((size_t)2 * L_SENT * G4 * 4);     // 8 MB
  _Float16* hcat_h = (_Float16*)alloc((size_t)L_SENT * 1024 * 2);       // 1 MB
  float*    feats  = (float*)   alloc((size_t)L_SENT * NTPAD * 4);      // 128 KB
  int*      bp     = (int*)     alloc((size_t)L_SENT * NTAG * 4);

  gather_embed_f16<<<(L_SENT * EMB + 255) / 256, 256, 0, stream>>>(sentence, embed, xh);
  prep_weights<<<(G4 * EMB + 255) / 256, 256, 0, stream>>>(
      Wih_f, Wih_b, bih_f, bhh_f, bih_b, bhh_b, W_out, wh, bsum, woh);

  dim3 ggrid(G4 / 512, L_SENT / 16, 2);          // 4 x 32 x 2 blocks of 8 waves
  gemm_xg_kernel<<<ggrid, 256, 0, stream>>>(xh, wh, bsum, xg);

  lstm_kernel<<<2, 512, 0, stream>>>(xg, Whh_f, Whh_b, hcat_h);

  dim3 fgrid(NTPAD / 16, L_SENT / 16);           // 4 x 32 single-wave tiles
  feats_wmma_kernel<<<fgrid, 32, 0, stream>>>(hcat_h, woh, b_out, feats);

  viterbi_kernel<<<1, 64, 0, stream>>>(feats, trans, bp, (float*)d_out);
}
